// QueryAttention_73555609911887
// MI455X (gfx1250) — compile-verified, matrix operand order checked by reading, not run on hardware
//
#include <hip/hip_runtime.h>
#include <math.h>
#include <stdint.h>

#define DEV __device__ __forceinline__

typedef __attribute__((ext_vector_type(16))) _Float16 v16h;
typedef __attribute__((ext_vector_type(8)))  _Float16 v8h;
typedef __attribute__((ext_vector_type(8)))  float    v8f;
typedef __attribute__((ext_vector_type(4)))  unsigned int v4u;
typedef __attribute__((ext_vector_type(8)))  int      v8i;
typedef __attribute__((ext_vector_type(4)))  int      v4i;

constexpr int Tn = 2048;   // sequence length
constexpr int Bn = 4;      // batch
constexpr int Dn = 1024;   // feature dim
constexpr int En = 2 * Dn; // kv projection dim

// ---------------------------------------------------------------------------
// CDNA5 async global->LDS copy (16B per lane), tracked by ASYNCcnt.
// ---------------------------------------------------------------------------
DEV void async_ld_b128(uint32_t lds_off, const void* gaddr) {
  asm volatile("global_load_async_to_lds_b128 %0, %1, off"
               :: "v"(lds_off), "v"(gaddr)
               : "memory");
}
DEV void wait_asynccnt0() {
  asm volatile("s_wait_asynccnt 0x0" ::: "memory");
}

// ---------------------------------------------------------------------------
// CDNA5 Tensor Data Mover: DMA a 2D tile (W elems of 2B per row, H rows,
// row stride ld elems) from global into LDS, packed row-major (W*2B rows).
// D# built per ISA 8.3/8.4; groups 2/3 zero (2D tensor). Tracked: TENSORcnt.
// ---------------------------------------------------------------------------
#if __has_builtin(__builtin_amdgcn_tensor_load_to_lds)
#define HAVE_TDM 1
DEV void tdm_load_2d(uint32_t lds_off, const void* gaddr, uint32_t W,
                     uint32_t H, uint32_t ld_elems) {
  uint64_t ga = (uint64_t)(uintptr_t)gaddr;
  v4u g0;
  g0[0] = 1u;                  // count=1, is_restore=0, gather off
  g0[1] = lds_off;             // lds_addr
  g0[2] = (uint32_t)ga;        // global_addr[31:0]
  g0[3] = (uint32_t)((ga >> 32) & 0x01FFFFFFu) | 0x80000000u;  // addr + type=2
  v8i g1 = {};
  g1[0] = (int)(1u << 16);                                   // data_size=2B
  g1[1] = (int)((W & 0xFFFFu) << 16);                        // tensor_dim0 lo
  g1[2] = (int)(((W >> 16) & 0xFFFFu) | ((H & 0xFFFFu) << 16));  // td0 hi, td1 lo
  g1[3] = (int)(((H >> 16) & 0xFFFFu) | ((W & 0xFFFFu) << 16));  // td1 hi, tile_dim0
  g1[4] = (int)(H & 0xFFFFu);                                // tile_dim1 (tile_dim2=0)
  g1[5] = (int)ld_elems;                                     // dim0_stride[31:0]
  v4i z = {};
#if defined(__clang_major__) && (__clang_major__ >= 23)
  v8i z8 = {};
  __builtin_amdgcn_tensor_load_to_lds(g0, g1, z, z, z8, 0);
#else
  __builtin_amdgcn_tensor_load_to_lds(g0, g1, z, z, 0);
#endif
}
#else
#define HAVE_TDM 0
#endif

DEV v16h cat8(v8h lo, v8h hi) {
  return __builtin_shufflevector(lo, hi, 0, 1, 2, 3, 4, 5, 6, 7, 8, 9, 10, 11,
                                 12, 13, 14, 15);
}

DEV v8f wmma_f16(v16h a, v16h b, v8f c) {
  return __builtin_amdgcn_wmma_f32_16x16x32_f16(false, a, false, b, (short)0,
                                                c, false, false);
}

// A/B fragment direct from global f16 (row-major, contiguous K).
DEV v16h ld_frag_h(const _Float16* __restrict__ base, int ld, int r0, int k0,
                   int lane) {
  int r  = r0 + (lane & 15);
  int kb = k0 + ((lane >> 4) << 3);
  const _Float16* p = base + (size_t)r * ld + kb;
  v8h lo = *reinterpret_cast<const v8h*>(p);
  v8h hi = *reinterpret_cast<const v8h*>(p + 16);
  __builtin_prefetch(p + 32, 0, 0);  // next K tile -> global_prefetch_b8
  return cat8(lo, hi);
}

// Same fragment, f32 source converted to f16 in registers.
DEV v16h ld_frag_f32(const float* __restrict__ base, int ld, int r0, int k0,
                     int lane) {
  int r  = r0 + (lane & 15);
  int kb = k0 + ((lane >> 4) << 3);
  const float* p = base + (size_t)r * ld + kb;
  __builtin_prefetch(p + 32, 0, 0);
  v16h out;
#pragma unroll
  for (int i = 0; i < 8; ++i) {
    out[i]     = (_Float16)p[i];
    out[i + 8] = (_Float16)p[16 + i];
  }
  return out;
}

// Fragment from an LDS tile with 32-half (64B) row stride.
DEV v16h ld_frag_lds(const _Float16* s, int row, int lane) {
  int r  = row + (lane & 15);
  int kb = (lane >> 4) << 3;
  v8h lo = *reinterpret_cast<const v8h*>(s + r * 32 + kb);
  v8h hi = *reinterpret_cast<const v8h*>(s + r * 32 + kb + 16);
  return cat8(lo, hi);
}

// ---------------------------------------------------------------------------
// q16[b][t][d] = query[t][b][d] * scale   (scale = D^-0.5 = 1/32)
// ---------------------------------------------------------------------------
__global__ __launch_bounds__(256) void cast_q_kernel(
    const float* __restrict__ query, _Float16* __restrict__ q16, float scale) {
  size_t i = (size_t)blockIdx.x * blockDim.x + threadIdx.x;  // (t,b,d) linear
  int d   = (int)(i % Dn);
  size_t tb = i / Dn;
  int b   = (int)(tb % Bn);
  size_t t = tb / Bn;
  q16[((size_t)b * Tn + t) * Dn + d] = (_Float16)(query[i] * scale);
}

// ---------------------------------------------------------------------------
// kv = X(8192x1024) * Win^T + bias ; K16 (B,T,D), V transposed Vt16 (B,D,T).
// Each wave: one 16-row M tile x two 16-col N tiles (A frag reused).
// ---------------------------------------------------------------------------
__global__ __launch_bounds__(128) void kv_proj_kernel(
    const float* __restrict__ X, const float* __restrict__ Win,
    const float* __restrict__ bin, _Float16* __restrict__ K16,
    _Float16* __restrict__ Vt16) {
  int lane = threadIdx.x & 31, wv = threadIdx.x >> 5;
  int r0 = (blockIdx.y * 4 + wv) * 16;  // row in flattened (t*B+b)
  int c0 = blockIdx.x * 32;             // e column (2 tiles)
  v8f acc0 = {}, acc1 = {};
#pragma unroll 2
  for (int k = 0; k < Dn; k += 32) {
    v16h a  = ld_frag_f32(X, Dn, r0, k, lane);
    v16h w0 = ld_frag_f32(Win, Dn, c0, k, lane);
    v16h w1 = ld_frag_f32(Win, Dn, c0 + 16, k, lane);
    acc0 = wmma_f16(a, w0, acc0);
    acc1 = wmma_f16(a, w1, acc1);
  }
  int e0 = c0 + (lane & 15), e1 = e0 + 16;
  float b0 = bin[e0], b1 = bin[e1];
  int mb = r0 + ((lane >> 4) << 3);
  if (c0 < Dn) {  // K half (block-uniform: 32-wide tile never straddles Dn)
#pragma unroll
    for (int r = 0; r < 8; ++r) {
      int m = mb + r, t = m >> 2, bb = m & 3;
      _Float16* row = K16 + ((size_t)bb * Tn + t) * Dn;
      row[e0] = (_Float16)(acc0[r] + b0);
      row[e1] = (_Float16)(acc1[r] + b1);
    }
  } else {  // V half, store transposed (d-major)
#pragma unroll
    for (int r = 0; r < 8; ++r) {
      int m = mb + r, t = m >> 2, bb = m & 3;
      Vt16[((size_t)bb * Dn + (e0 - Dn)) * Tn + t] = (_Float16)(acc0[r] + b0);
      Vt16[((size_t)bb * Dn + (e1 - Dn)) * Tn + t] = (_Float16)(acc1[r] + b1);
    }
  }
}

// ---------------------------------------------------------------------------
// logits[b][t][s] = sum_d Q16[b][t][d] * K16[b][s][d]
// Block (128 thr) computes 64(M) x 64(N). Per K-step, wave 0 issues two TDM
// 2D-tile descriptors (A 64x32, B 64x32 halves) -> LDS, waits TENSORcnt.
// Each wave: 1 M tile x 4 N tiles -> 4 v_wmma per K-step, A frag reused 4x.
// ---------------------------------------------------------------------------
__global__ __launch_bounds__(128) void qk_kernel(
    const _Float16* __restrict__ Q16, const _Float16* __restrict__ K16,
    float* __restrict__ Wl) {
  __shared__ __align__(16) _Float16 sA[64 * 32];
  __shared__ __align__(16) _Float16 sB[64 * 32];
  int bz = blockIdx.z;
  const _Float16* A  = Q16 + (size_t)bz * Tn * Dn;
  const _Float16* Bm = K16 + (size_t)bz * Tn * Dn;
  float* C = Wl + (size_t)bz * Tn * Tn;
  int tid = threadIdx.x, lane = tid & 31, wv = tid >> 5;
  int m0 = blockIdx.y * 64;
  int n0 = blockIdx.x * 64;
  uint32_t sAo = (uint32_t)(uintptr_t)sA;
  uint32_t sBo = (uint32_t)(uintptr_t)sB;
#if !HAVE_TDM
  int a0r = (tid * 2) >> 2, a0c = (tid * 2) & 3;
  int a1r = (tid * 2 + 1) >> 2, a1c = (tid * 2 + 1) & 3;
#endif
  v8f acc[4] = {{}, {}, {}, {}};
  for (int k = 0; k < Dn; k += 32) {
#if HAVE_TDM
    if (wv == 0) {
      tdm_load_2d(sAo, A + (size_t)m0 * Dn + k, 32, 64, Dn);
      tdm_load_2d(sBo, Bm + (size_t)n0 * Dn + k, 32, 64, Dn);
      __builtin_amdgcn_s_wait_tensorcnt(0);
    }
    __syncthreads();
#else
    async_ld_b128(sAo + (uint32_t)(a0r * 64 + a0c * 16),
                  A + (size_t)(m0 + a0r) * Dn + k + a0c * 8);
    async_ld_b128(sAo + (uint32_t)(a1r * 64 + a1c * 16),
                  A + (size_t)(m0 + a1r) * Dn + k + a1c * 8);
    async_ld_b128(sBo + (uint32_t)(a0r * 64 + a0c * 16),
                  Bm + (size_t)(n0 + a0r) * Dn + k + a0c * 8);
    async_ld_b128(sBo + (uint32_t)(a1r * 64 + a1c * 16),
                  Bm + (size_t)(n0 + a1r) * Dn + k + a1c * 8);
    wait_asynccnt0();
    __syncthreads();
#endif
    v16h a = ld_frag_lds(sA, wv * 16, lane);
#pragma unroll
    for (int j = 0; j < 4; ++j) {
      v16h b = ld_frag_lds(sB, j * 16, lane);
      acc[j] = wmma_f16(a, b, acc[j]);
    }
    __syncthreads();
  }
  int n = lane & 15;
  int mb = m0 + wv * 16 + ((lane >> 4) << 3);
#pragma unroll
  for (int r = 0; r < 8; ++r) {
    float* row = C + (size_t)(mb + r) * Tn + n0;
#pragma unroll
    for (int j = 0; j < 4; ++j) row[j * 16 + n] = acc[j][r];
  }
}

// ---------------------------------------------------------------------------
// Row softmax over s (length Tn), LDS tree reductions. Writes normalized f32
// in place (attn_weights output) and an f16 copy for S*V.
// ---------------------------------------------------------------------------
__global__ __launch_bounds__(256) void softmax_kernel(
    float* __restrict__ w, _Float16* __restrict__ w16) {
  __shared__ float red[256];
  size_t row = blockIdx.x;  // b*Tn + t
  float* p = w + row * Tn;
  int tid = threadIdx.x;
  float m = -INFINITY;
  for (int s = tid; s < Tn; s += 256) m = fmaxf(m, p[s]);
  red[tid] = m;
  __syncthreads();
  for (int off = 128; off > 0; off >>= 1) {
    if (tid < off) red[tid] = fmaxf(red[tid], red[tid + off]);
    __syncthreads();
  }
  m = red[0];
  __syncthreads();
  float sum = 0.f;
  for (int s = tid; s < Tn; s += 256) {
    float e = __expf(p[s] - m);
    p[s] = e;
    sum += e;
  }
  red[tid] = sum;
  __syncthreads();
  for (int off = 128; off > 0; off >>= 1) {
    if (tid < off) red[tid] += red[tid + off];
    __syncthreads();
  }
  float inv = 1.0f / red[0];
  _Float16* q = w16 + row * Tn;
  for (int s = tid; s < Tn; s += 256) {
    float v = p[s] * inv;
    p[s] = v;
    q[s] = (_Float16)v;
  }
}

// ---------------------------------------------------------------------------
// attn16[(t*B+b)][d] = sum_s W16[b][t][s] * Vt16[b][d][s]
// Block computes 64(t) x 64(d); async global->LDS staging (ASYNCcnt path).
// ---------------------------------------------------------------------------
__global__ __launch_bounds__(128) void av_kernel(
    const _Float16* __restrict__ W16, const _Float16* __restrict__ Vt16,
    _Float16* __restrict__ A16) {
  __shared__ __align__(16) _Float16 sA[64 * 32];
  __shared__ __align__(16) _Float16 sB[64 * 32];
  int bz = blockIdx.z;
  const _Float16* A  = W16 + (size_t)bz * Tn * Tn;   // (t, s)
  const _Float16* Bm = Vt16 + (size_t)bz * Dn * Tn;  // (d, s)
  int tid = threadIdx.x, lane = tid & 31, wv = tid >> 5;
  int m0 = blockIdx.y * 64;  // t
  int n0 = blockIdx.x * 64;  // d
  uint32_t sAo = (uint32_t)(uintptr_t)sA;
  uint32_t sBo = (uint32_t)(uintptr_t)sB;
  int a0r = (tid * 2) >> 2, a0c = (tid * 2) & 3;
  int a1r = (tid * 2 + 1) >> 2, a1c = (tid * 2 + 1) & 3;
  v8f acc[4] = {{}, {}, {}, {}};
  for (int k = 0; k < Tn; k += 32) {
    async_ld_b128(sAo + (uint32_t)(a0r * 64 + a0c * 16),
                  A + (size_t)(m0 + a0r) * Tn + k + a0c * 8);
    async_ld_b128(sAo + (uint32_t)(a1r * 64 + a1c * 16),
                  A + (size_t)(m0 + a1r) * Tn + k + a1c * 8);
    async_ld_b128(sBo + (uint32_t)(a0r * 64 + a0c * 16),
                  Bm + (size_t)(n0 + a0r) * Tn + k + a0c * 8);
    async_ld_b128(sBo + (uint32_t)(a1r * 64 + a1c * 16),
                  Bm + (size_t)(n0 + a1r) * Tn + k + a1c * 8);
    wait_asynccnt0();
    __syncthreads();
    v16h a = ld_frag_lds(sA, wv * 16, lane);
#pragma unroll
    for (int j = 0; j < 4; ++j) {
      v16h b = ld_frag_lds(sB, j * 16, lane);
      acc[j] = wmma_f16(a, b, acc[j]);
    }
    __syncthreads();
  }
  int n = lane & 15;
  int mb = m0 + wv * 16 + ((lane >> 4) << 3);
#pragma unroll
  for (int r = 0; r < 8; ++r) {
    int t = mb + r;
    _Float16* row = A16 + ((size_t)t * Bn + bz) * Dn + n0;
#pragma unroll
    for (int j = 0; j < 4; ++j) row[j * 16 + n] = (_Float16)acc[j][r];
  }
}

// ---------------------------------------------------------------------------
// out[m][e] = sum_d A16[m][d] * Wout[e][d] + bout[e]   (m = t*B+b)
// ---------------------------------------------------------------------------
__global__ __launch_bounds__(128) void out_proj_kernel(
    const _Float16* __restrict__ A16, const float* __restrict__ Wout,
    const float* __restrict__ bout, float* __restrict__ out) {
  int lane = threadIdx.x & 31, wv = threadIdx.x >> 5;
  int r0 = (blockIdx.y * 4 + wv) * 16;
  int c0 = blockIdx.x * 32;
  v8f acc0 = {}, acc1 = {};
#pragma unroll 2
  for (int k = 0; k < Dn; k += 32) {
    v16h a  = ld_frag_h(A16, Dn, r0, k, lane);
    v16h w0 = ld_frag_f32(Wout, Dn, c0, k, lane);
    v16h w1 = ld_frag_f32(Wout, Dn, c0 + 16, k, lane);
    acc0 = wmma_f16(a, w0, acc0);
    acc1 = wmma_f16(a, w1, acc1);
  }
  int e0 = c0 + (lane & 15), e1 = e0 + 16;
  float b0 = bout[e0], b1 = bout[e1];
  int mb = r0 + ((lane >> 4) << 3);
#pragma unroll
  for (int r = 0; r < 8; ++r) {
    float* row = out + (size_t)(mb + r) * Dn;
    row[e0] = acc0[r] + b0;
    row[e1] = acc1[r] + b1;
  }
}

// ---------------------------------------------------------------------------
extern "C" void kernel_launch(void* const* d_in, const int* in_sizes, int n_in,
                              void* d_out, int out_size, void* d_ws,
                              size_t ws_size, hipStream_t stream) {
  const float* query = (const float*)d_in[0];
  const float* xf    = (const float*)d_in[1];
  const float* win   = (const float*)d_in[2];
  const float* bin   = (const float*)d_in[3];
  const float* wout  = (const float*)d_in[4];
  const float* bout  = (const float*)d_in[5];

  float* out_attn = (float*)d_out;                    // (T,B,D) f32
  float* out_w    = out_attn + (size_t)Tn * Bn * Dn;  // (B,T,T) f32

  const size_t QB = (size_t)Bn * Tn * Dn;  // halves per buffer
  _Float16* Q16  = (_Float16*)d_ws;        // (B,T,D)
  _Float16* K16  = Q16 + QB;               // (B,T,D)
  _Float16* Vt16 = K16 + QB;               // (B,D,T)
  _Float16* A16  = Vt16 + QB;              // (T*B,D) attn intermediate
  _Float16* W16  = Q16;  // softmax f16 weights alias Q16+K16 (dead by then)
  (void)in_sizes; (void)n_in; (void)out_size; (void)ws_size;

  const float scale = 0.03125f;  // 1024^-0.5

  cast_q_kernel<<<(Tn * Bn * Dn) / 256, 256, 0, stream>>>(query, Q16, scale);
  kv_proj_kernel<<<dim3(En / 32, (Tn * Bn / 16) / 4), 128, 0, stream>>>(
      xf, win, bin, K16, Vt16);
  qk_kernel<<<dim3(Tn / 64, Tn / 64, Bn), 128, 0, stream>>>(Q16, K16, out_w);
  softmax_kernel<<<Bn * Tn, 256, 0, stream>>>(out_w, W16);
  av_kernel<<<dim3(Dn / 64, Tn / 64, Bn), 128, 0, stream>>>(W16, Vt16, A16);
  out_proj_kernel<<<dim3(Dn / 32, (Tn * Bn / 16) / 4), 128, 0, stream>>>(
      A16, wout, bout, out_attn);
}
